// MultiheadAttention_34505767256362
// MI455X (gfx1250) — compile-verified, matrix operand order checked by reading, not run on hardware
//
#include <hip/hip_runtime.h>
#include <hip/hip_bf16.h>

// Problem constants (from the reference).
#define NN    3072   // nodes
#define IND   512    // input dim
#define HD    512    // H*D
#define DH    64     // per-head dim
#define H7OFF 448    // column offset of head 7 (= 7*64)

typedef __attribute__((ext_vector_type(16))) __bf16 v16bf;
typedef __attribute__((ext_vector_type(8)))  __bf16 v8bf;
typedef __attribute__((ext_vector_type(8)))  float  v8f;

// f32 -> bf16 round-to-nearest-even (bit-exact, no reliance on cast lowering).
static __device__ __forceinline__ __bf16 f2bf(float f) {
    union { float f; unsigned u; } v; v.f = f;
    unsigned r = (v.u + 0x7FFFu + ((v.u >> 16) & 1u)) >> 16;
    unsigned short s = (unsigned short)r;
    __bf16 out;
    __builtin_memcpy(&out, &s, sizeof(out));
    return out;
}

// Load one 16-half WMMA fragment: two contiguous 16-byte chunks
// (elements {0..7} at p, {8..15} at p+16 halves) -> two global_load_b128.
static __device__ __forceinline__ v16bf load_frag(const __bf16* __restrict__ p) {
    v8bf lo = *(const v8bf*)(p);
    v8bf hi = *(const v8bf*)(p + 16);
    return __builtin_shufflevector(lo, hi, 0, 1, 2, 3, 4, 5, 6, 7,
                                           8, 9, 10, 11, 12, 13, 14, 15);
}

// One-time f32 -> bf16 conversion of X (row-major, layout preserved).
__global__ void cvt_x_kernel(const float* __restrict__ X, __bf16* __restrict__ Xb)
{
    int i = blockIdx.x * blockDim.x + threadIdx.x;
    if (i < NN * IND) Xb[i] = f2bf(X[i]);
}

// One-time f32 -> bf16 + transpose of a 512x512 W: WT[c*512 + k] = W[k*512 + c].
// Consecutive threads write consecutive k -> coalesced stores.
__global__ void cvt_wT_kernel(const float* __restrict__ W, __bf16* __restrict__ WT)
{
    int i = blockIdx.x * blockDim.x + threadIdx.x;
    if (i >= IND * HD) return;
    int c = i >> 9, k = i & 511;
    WT[i] = f2bf(W[k * HD + c]);
}

// ---------------------------------------------------------------------------
// QKV projection: Out = X(3072x512) @ W(512x512) + b, one of {Q,K,V} per
// blockIdx.z. 8 waves/block as 2x4; each wave owns a 32x64 C tile
// (2x4 v_wmma_f32_16x16x32_bf16 accumulators). Block tile = 64 x 256.
// A and B(=WT) are pre-converted bf16; all fragment loads are b128.
// ---------------------------------------------------------------------------
__global__ __launch_bounds__(256)
void qkv_gemm_wmma(const __bf16* __restrict__ Xb,
                   const __bf16* __restrict__ WT0, const __bf16* __restrict__ WT1,
                   const __bf16* __restrict__ WT2,
                   const float* __restrict__ b0, const float* __restrict__ b1,
                   const float* __restrict__ b2,
                   float* __restrict__ Q, float* __restrict__ Kp,
                   float* __restrict__ Vp)
{
    const __bf16* WT; const float* bias; float* Out;
    if (blockIdx.z == 0)      { WT = WT0; bias = b0; Out = Q;  }
    else if (blockIdx.z == 1) { WT = WT1; bias = b1; Out = Kp; }
    else                      { WT = WT2; bias = b2; Out = Vp; }

    const int lane = threadIdx.x & 31;
    const int wid  = threadIdx.x >> 5;
    const int wr   = wid >> 2;      // 0..1 : wave row in block
    const int wc   = wid & 3;       // 0..3 : wave col in block
    const int h    = lane >> 4;     // lane half (K-group select per ISA layout)
    const int r    = lane & 15;     // row (A) / col (B,C) within 16

    const int rowBase = blockIdx.y * 64  + wr * 32;
    const int colBase = blockIdx.x * 256 + wc * 64;

    const __bf16* aBase0 = Xb + (size_t)(rowBase + r) * IND + 8 * h;
    const __bf16* aBase1 = aBase0 + (size_t)16 * IND;
    const __bf16* bBase  = WT + (size_t)(colBase + r) * IND + 8 * h;

    v8f acc[2][4] = {};

    for (int kk = 0; kk < IND; kk += 32) {
        v16bf a[2], b[4];
        a[0] = load_frag(aBase0 + kk);
        a[1] = load_frag(aBase1 + kk);
        #pragma unroll
        for (int nt = 0; nt < 4; ++nt)
            b[nt] = load_frag(bBase + (size_t)(nt * 16) * IND + kk);
        #pragma unroll
        for (int mt = 0; mt < 2; ++mt)
            #pragma unroll
            for (int nt = 0; nt < 4; ++nt)
                acc[mt][nt] = __builtin_amdgcn_wmma_f32_16x16x32_bf16(
                    false, a[mt], false, b[nt], (short)0, acc[mt][nt],
                    false, false);
    }

    // C/D layout: VGPR j -> row (8h + j), col r.
    #pragma unroll
    for (int mt = 0; mt < 2; ++mt) {
        #pragma unroll
        for (int nt = 0; nt < 4; ++nt) {
            const int col = colBase + nt * 16 + r;
            const float bb = bias[col];
            #pragma unroll
            for (int j = 0; j < 8; ++j) {
                const int row = rowBase + mt * 16 + 8 * h + j;
                Out[(size_t)row * HD + col] = acc[mt][nt][j] + bb;
            }
        }
    }
}

// Zero the head-7 accumulators and row denominators.
__global__ void zero_kernel(float* __restrict__ acc7, float* __restrict__ rowAdd)
{
    int i = blockIdx.x * blockDim.x + threadIdx.x;
    if (i < NN * DH) acc7[i] = 0.0f;
    if (i < NN)      rowAdd[i] = 0.0f;
}

// Column sums of V (all 512 columns); heads 0-6 need them /N, head 7 as base.
__global__ void colsum_kernel(const float* __restrict__ V, float* __restrict__ vsum)
{
    int c = blockIdx.x * blockDim.x + threadIdx.x;
    if (c >= HD) return;
    float s = 0.0f;
    for (int n = 0; n < NN; ++n) s += V[(size_t)n * HD + c];
    vsum[c] = s;
}

// Heads 0..6: uniform softmax -> column mean of V broadcast to every node.
__global__ void bcast_kernel(const float* __restrict__ vsum, float* __restrict__ out)
{
    int i = blockIdx.x * blockDim.x + threadIdx.x;
    if (i >= NN * H7OFF) return;
    int n = i / H7OFF, c = i % H7OFF;
    out[(size_t)n * HD + c] = vsum[c] * (1.0f / (float)NN);
}

// One wave32 per edge: s = q7[src].k7[dst]/8, w = e^s - 1;
// rowAdd[src] += w; acc7[src,:] += w * v7[dst,:].
__global__ __launch_bounds__(256)
void edge_kernel(const int* __restrict__ ei, int E,
                 const float* __restrict__ Q, const float* __restrict__ Kp,
                 const float* __restrict__ Vp,
                 float* __restrict__ acc7, float* __restrict__ rowAdd)
{
    const int lane = threadIdx.x & 31;
    const int e = blockIdx.x * 8 + (threadIdx.x >> 5);
    if (e >= E) return;
    const int src = ei[e];
    const int dst = ei[E + e];
    const float* q = Q  + (size_t)src * HD + H7OFF;
    const float* k = Kp + (size_t)dst * HD + H7OFF;
    const float* v = Vp + (size_t)dst * HD + H7OFF;
    const int d0 = lane * 2;
    float p = q[d0] * k[d0] + q[d0 + 1] * k[d0 + 1];
    #pragma unroll
    for (int m = 16; m > 0; m >>= 1) p += __shfl_xor(p, m, 32);
    const float w = __expf(p * 0.125f) - 1.0f;   // 1/sqrt(64) = 0.125
    atomicAdd(&acc7[(size_t)src * DH + d0],     w * v[d0]);
    atomicAdd(&acc7[(size_t)src * DH + d0 + 1], w * v[d0 + 1]);
    if (lane == 0) atomicAdd(&rowAdd[src], w);
}

// Head 7: out = (sum_m v7 + edge-weighted part) / (N + sum_edges (e^s - 1)).
__global__ void finalize_kernel(const float* __restrict__ vsum,
                                const float* __restrict__ acc7,
                                const float* __restrict__ rowAdd,
                                float* __restrict__ out)
{
    int i = blockIdx.x * blockDim.x + threadIdx.x;
    if (i >= NN * DH) return;
    int n = i / DH, d = i % DH;
    out[(size_t)n * HD + H7OFF + d] =
        (vsum[H7OFF + d] + acc7[i]) / ((float)NN + rowAdd[n]);
}

extern "C" void kernel_launch(void* const* d_in, const int* in_sizes, int n_in,
                              void* d_out, int out_size, void* d_ws, size_t ws_size,
                              hipStream_t stream)
{
    const float* x  = (const float*)d_in[0];
    const int*   ei = (const int*)  d_in[1];  // [2,E] flattened: src row, dst row
    const float* Wq = (const float*)d_in[2];
    const float* Wk = (const float*)d_in[3];
    const float* Wv = (const float*)d_in[4];
    const float* bq = (const float*)d_in[5];
    const float* bk = (const float*)d_in[6];
    const float* bv = (const float*)d_in[7];
    float* out = (float*)d_out;
    const int E = in_sizes[1] / 2;

    // Workspace layout:
    //   floats:  Q | K | V | vsum(512) | rowAdd(N) | acc7(N*64)
    //   bf16  :  Xb(N*512) | WTq | WTk | WTv (each 512*512)
    float* ws     = (float*)d_ws;
    float* Q      = ws;
    float* K      = Q + (size_t)NN * HD;
    float* V      = K + (size_t)NN * HD;
    float* vsum   = V + (size_t)NN * HD;
    float* rowAdd = vsum + HD;
    float* acc7   = rowAdd + NN;
    __bf16* Xb    = (__bf16*)(acc7 + (size_t)NN * DH);
    __bf16* WTq   = Xb  + (size_t)NN * IND;
    __bf16* WTk   = WTq + (size_t)IND * HD;
    __bf16* WTv   = WTk + (size_t)IND * HD;

    // One-time precision conversion (bf16 A/B, f32 accumulate downstream).
    cvt_x_kernel<<<(NN * IND + 255) / 256, 256, 0, stream>>>(x, Xb);
    cvt_wT_kernel<<<(IND * HD + 255) / 256, 256, 0, stream>>>(Wq, WTq);
    cvt_wT_kernel<<<(IND * HD + 255) / 256, 256, 0, stream>>>(Wk, WTk);
    cvt_wT_kernel<<<(IND * HD + 255) / 256, 256, 0, stream>>>(Wv, WTv);
    zero_kernel<<<(NN * DH + 255) / 256, 256, 0, stream>>>(acc7, rowAdd);

    dim3 g(HD / 256, NN / 64, 3);
    qkv_gemm_wmma<<<g, 256, 0, stream>>>(Xb, WTq, WTk, WTv, bq, bk, bv, Q, K, V);

    colsum_kernel<<<(HD + 255) / 256, 256, 0, stream>>>(V, vsum);
    bcast_kernel<<<(NN * H7OFF + 255) / 256, 256, 0, stream>>>(vsum, out);
    edge_kernel<<<(E + 7) / 8, 256, 0, stream>>>(ei, E, Q, K, V, acc7, rowAdd);
    finalize_kernel<<<(NN * DH + 255) / 256, 256, 0, stream>>>(vsum, acc7, rowAdd, out);
}
